// GraphSAGENet_35897336660097
// MI455X (gfx1250) — matrix operand order repaired and checked by reading, not compiled, not run on hardware
//
#include <hip/hip_runtime.h>

// ---------------- problem constants ----------------
constexpr int N_NODES = 100000;
constexpr int E_EDGES = 1600000;
constexpr int IN_C    = 128;
constexpr int HID     = 128;
constexpr int CLS     = 40;
constexpr int CLS_PAD = 48;           // 3 tiles of 16
constexpr int TILES   = N_NODES / 16; // 6250, exact
constexpr int WAVES_PER_BLK = 8;
constexpr int BLOCKS_GEMM   = (TILES + WAVES_PER_BLK - 1) / WAVES_PER_BLK; // 782

typedef __attribute__((ext_vector_type(16))) __bf16 v16bf;
typedef __attribute__((ext_vector_type(8)))  __bf16 v8bf;
typedef __attribute__((ext_vector_type(4)))  __bf16 v4bf;
typedef __attribute__((ext_vector_type(2)))  __bf16 v2bf;
typedef __attribute__((ext_vector_type(8)))  float  v8f;

// f32 -> bf16 round-to-nearest-even
__device__ inline __bf16 f2bf(float f) {
    unsigned u = __builtin_bit_cast(unsigned, f);
    unsigned r = (u + 0x7FFFu + ((u >> 16) & 1u)) >> 16;
    unsigned short s = (unsigned short)r;
    return __builtin_bit_cast(__bf16, s);
}
// bf16 -> f32 (exact)
__device__ inline float bf2f(__bf16 b) {
    unsigned u = ((unsigned)__builtin_bit_cast(unsigned short, b)) << 16;
    return __builtin_bit_cast(float, u);
}

// ---------------- utility kernels ----------------
__global__ void zero_kernel(float* __restrict__ p, long long n) {
    long long i = (long long)blockIdx.x * blockDim.x + threadIdx.x;
    if (i < n) p[i] = 0.0f;
}

__global__ void degree_kernel(const long long* __restrict__ dst,
                              float* __restrict__ cnt, int E) {
    int e = blockIdx.x * blockDim.x + threadIdx.x;
    if (e < E) atomicAdd(&cnt[(int)dst[e]], 1.0f);
}

// streaming f32 -> bf16 (4 elements / thread, b64 store)
__global__ void convert_x_kernel(const float* __restrict__ in,
                                 __bf16* __restrict__ out, long long n) {
    long long i4 = ((long long)blockIdx.x * blockDim.x + threadIdx.x) * 4;
    if (i4 >= n) return;
    const float4 s = *(const float4*)(in + i4);
    v4bf o;
    o[0] = f2bf(s.x); o[1] = f2bf(s.y); o[2] = f2bf(s.z); o[3] = f2bf(s.w);
    *(v4bf*)(out + i4) = o;
}

// streaming agg = sum * (1/max(cnt,1)) -> bf16 (4 elements / thread, one row)
__global__ void convert_agg_kernel(const float* __restrict__ sum,
                                   const float* __restrict__ cnt,
                                   __bf16* __restrict__ agg, long long n) {
    long long i4 = ((long long)blockIdx.x * blockDim.x + threadIdx.x) * 4;
    if (i4 >= n) return;
    const int node = (int)(i4 >> 7);                 // 128 cols per node
    const float inv = 1.0f / fmaxf(cnt[node], 1.0f);
    const float4 s = *(const float4*)(sum + i4);
    v4bf o;
    o[0] = f2bf(s.x * inv); o[1] = f2bf(s.y * inv);
    o[2] = f2bf(s.z * inv); o[3] = f2bf(s.w * inv);
    *(v4bf*)(agg + i4) = o;
}

// edge scatter-add: bf16 feature gather (b128 = 8 cols), f32 atomics accumulate
__global__ void scatter_bf16_kernel(const __bf16* __restrict__ feat,
                                    const long long* __restrict__ src,
                                    const long long* __restrict__ dst,
                                    float* __restrict__ sum, int E) {
    long long t = (long long)blockIdx.x * blockDim.x + threadIdx.x;
    if (t >= (long long)E * 16) return;
    const int e  = (int)(t >> 4);
    const int c8 = (int)(t & 15) * 8;
    const int s  = (int)src[e];
    const int d  = (int)dst[e];
    const v8bf v = *(const v8bf*)(feat + (long long)s * 128 + c8);
    float* p = sum + (long long)d * 128 + c8;
    #pragma unroll
    for (int i = 0; i < 8; ++i)
        atomicAdd(p + i, bf2f(v[i]));
}

// ---------------- layer 1: h = relu(agg @ W1_l^T + b1 + x @ W1_r^T) ----------------
__global__ __launch_bounds__(256) void sage_layer1(
    const __bf16* __restrict__ aggbf, const __bf16* __restrict__ xbf,
    const float* __restrict__ W1l, const float* __restrict__ b1,
    const float* __restrict__ W1r,
    __bf16* __restrict__ hbf)
{
    __shared__ alignas(32) __bf16 ldsWl[HID * IN_C];   // 32 KB
    __shared__ alignas(32) __bf16 ldsWr[HID * IN_C];   // 32 KB

    const int tid = threadIdx.x;
    for (int i = tid * 2; i < HID * IN_C; i += 512) {
        const float2 wl = *(const float2*)(W1l + i);
        const float2 wr = *(const float2*)(W1r + i);
        v2bf l; l[0] = f2bf(wl.x); l[1] = f2bf(wl.y);
        v2bf r; r[0] = f2bf(wr.x); r[1] = f2bf(wr.y);
        *(v2bf*)(ldsWl + i) = l;
        *(v2bf*)(ldsWr + i) = r;
    }
    __syncthreads();

    const int wave = tid >> 5;
    const int lane = tid & 31;
    const int tileRaw = blockIdx.x * WAVES_PER_BLK + wave;
    const bool active = (tileRaw < TILES);
    const int tile = active ? tileRaw : 0;

    const int m     = lane & 15;            // A-row within tile / B,C col within tile
    const int row   = tile * 16 + m;        // global row this lane loads for A
    const int khalf = (lane >> 4) * 16;     // A K-half: contiguous 16 per lane half

    const __bf16* aggRow = aggbf + (long long)row * 128;
    const __bf16* xRow   = xbf   + (long long)row * 128;

    v8f acc[8];
    #pragma unroll
    for (int j = 0; j < 8; ++j)
        #pragma unroll
        for (int e = 0; e < 8; ++e) acc[j][e] = 0.0f;

    #pragma unroll
    for (int kc = 0; kc < 4; ++kc) {
        const int kb = kc * 32;
        // A fragments: two contiguous 8-bf16 runs per lane -> b128 loads, no conversion
        union { v16bf v; v8bf h[2]; } aAgg, aX;
        #pragma unroll
        for (int half = 0; half < 2; ++half) {
            const int kstart = kb + khalf + 8 * half;
            aAgg.h[half] = *(const v8bf*)(aggRow + kstart);
            aX.h[half]   = *(const v8bf*)(xRow + kstart);
        }
        const int kB = kb + ((lane >= 16) ? 16 : 0);
        #pragma unroll
        for (int j = 0; j < 8; ++j) {
            const int colg = j * 16 + m;     // output channel (weight row)
            const v16bf bL = *(const v16bf*)(ldsWl + colg * 128 + kB);
            const v16bf bR = *(const v16bf*)(ldsWr + colg * 128 + kB);
            acc[j] = __builtin_amdgcn_wmma_f32_16x16x32_bf16(
                false, aAgg.v, false, bL, (short)0, acc[j], false, false);
            acc[j] = __builtin_amdgcn_wmma_f32_16x16x32_bf16(
                false, aX.v,   false, bR, (short)0, acc[j], false, false);
        }
    }

    if (active) {
        const int col   = lane & 15;
        const int rbase = tile * 16 + ((lane < 16) ? 0 : 8);
        #pragma unroll
        for (int j = 0; j < 8; ++j) {
            const int cIdx = j * 16 + col;
            const float bias = b1[cIdx];
            #pragma unroll
            for (int r = 0; r < 8; ++r) {
                const float v = fmaxf(acc[j][r] + bias, 0.0f);
                hbf[(long long)(rbase + r) * 128 + cIdx] = f2bf(v);
            }
        }
    }
}

// ---------------- layer 2: log_softmax(agg2 @ W2_l^T + b2 + h @ W2_r^T) ----------------
__global__ __launch_bounds__(256) void sage_layer2(
    const __bf16* __restrict__ aggbf, const __bf16* __restrict__ hbf,
    const float* __restrict__ W2l, const float* __restrict__ b2,
    const float* __restrict__ W2r,
    float* __restrict__ out)
{
    __shared__ alignas(32) __bf16 ldsWl[CLS_PAD * HID];       // 12 KB
    __shared__ alignas(32) __bf16 ldsWr[CLS_PAD * HID];       // 12 KB
    __shared__ float  ldsLogit[WAVES_PER_BLK][16 * CLS_PAD];  // 24 KB

    const int tid = threadIdx.x;
    for (int i = tid * 2; i < CLS_PAD * HID; i += 512) {
        const int r = i / HID;       // pairs never straddle rows (HID even)
        float a = 0.0f, b = 0.0f, c = 0.0f, d = 0.0f;
        if (r < CLS) {
            const float2 wl = *(const float2*)(W2l + (size_t)r * HID + (i % HID));
            const float2 wr = *(const float2*)(W2r + (size_t)r * HID + (i % HID));
            a = wl.x; b = wl.y; c = wr.x; d = wr.y;
        }
        v2bf l; l[0] = f2bf(a); l[1] = f2bf(b);
        v2bf rr; rr[0] = f2bf(c); rr[1] = f2bf(d);
        *(v2bf*)(ldsWl + i) = l;
        *(v2bf*)(ldsWr + i) = rr;
    }
    __syncthreads();

    const int wave = tid >> 5;
    const int lane = tid & 31;
    const int tileRaw = blockIdx.x * WAVES_PER_BLK + wave;
    const bool active = (tileRaw < TILES);
    const int tile = active ? tileRaw : 0;

    const int m     = lane & 15;
    const int row   = tile * 16 + m;
    const int khalf = (lane >> 4) * 16;

    const __bf16* aggRow = aggbf + (long long)row * 128;
    const __bf16* hRow   = hbf   + (long long)row * 128;

    v8f acc[3];
    #pragma unroll
    for (int j = 0; j < 3; ++j)
        #pragma unroll
        for (int e = 0; e < 8; ++e) acc[j][e] = 0.0f;

    #pragma unroll
    for (int kc = 0; kc < 4; ++kc) {
        const int kb = kc * 32;
        union { v16bf v; v8bf h[2]; } aAgg, aH;
        #pragma unroll
        for (int half = 0; half < 2; ++half) {
            const int kstart = kb + khalf + 8 * half;
            aAgg.h[half] = *(const v8bf*)(aggRow + kstart);
            aH.h[half]   = *(const v8bf*)(hRow + kstart);
        }
        const int kB = kb + ((lane >= 16) ? 16 : 0);
        #pragma unroll
        for (int j = 0; j < 3; ++j) {
            const int colg = j * 16 + m;
            const v16bf bL = *(const v16bf*)(ldsWl + colg * 128 + kB);
            const v16bf bR = *(const v16bf*)(ldsWr + colg * 128 + kB);
            acc[j] = __builtin_amdgcn_wmma_f32_16x16x32_bf16(
                false, aAgg.v, false, bL, (short)0, acc[j], false, false);
            acc[j] = __builtin_amdgcn_wmma_f32_16x16x32_bf16(
                false, aH.v,   false, bR, (short)0, acc[j], false, false);
        }
    }

    // stage logits (with bias) into this wave's LDS slab
    {
        const int col  = lane & 15;
        const int rOff = (lane < 16) ? 0 : 8;
        #pragma unroll
        for (int j = 0; j < 3; ++j) {
            const int cIdx = j * 16 + col;
            const float bias = (cIdx < CLS) ? b2[cIdx] : 0.0f;
            #pragma unroll
            for (int r = 0; r < 8; ++r)
                ldsLogit[wave][(rOff + r) * CLS_PAD + cIdx] = acc[j][r] + bias;
        }
    }
    __syncthreads();   // uniform: every thread reaches here

    // lanes 0-15: one output row each -> 40-class log_softmax
    if (active && lane < 16) {
        const float* rowp = &ldsLogit[wave][(lane & 15) * CLS_PAD];
        float mx = -3.0e38f;
        #pragma unroll
        for (int c = 0; c < CLS; ++c) mx = fmaxf(mx, rowp[c]);
        float se = 0.0f;
        #pragma unroll
        for (int c = 0; c < CLS; ++c) se += __expf(rowp[c] - mx);
        const float lse = mx + __logf(se);
        const int gRow = tile * 16 + (lane & 15);
        #pragma unroll
        for (int c = 0; c < CLS; ++c)
            out[gRow * CLS + c] = rowp[c] - lse;
    }
}

// ---------------- launcher ----------------
extern "C" void kernel_launch(void* const* d_in, const int* in_sizes, int n_in,
                              void* d_out, int out_size, void* d_ws, size_t ws_size,
                              hipStream_t stream) {
    const float*     x    = (const float*)d_in[0];
    const long long* ei   = (const long long*)d_in[1];
    const long long* src  = ei;                 // edge_index[0]
    const long long* dst  = ei + E_EDGES;       // edge_index[1]
    const float*     W1l  = (const float*)d_in[2];
    const float*     b1   = (const float*)d_in[3];
    const float*     W1r  = (const float*)d_in[4];
    const float*     W2l  = (const float*)d_in[5];
    const float*     b2   = (const float*)d_in[6];
    const float*     W2r  = (const float*)d_in[7];
    float*           out  = (float*)d_out;

    const long long featN = (long long)N_NODES * 128;

    // workspace: cnt[N] f32 | sum[N*128] f32 | xbf | hbf | aggbf (bf16) ~128 MB
    float*  cnt   = (float*)d_ws;
    float*  sum   = cnt + N_NODES;
    __bf16* xbf   = (__bf16*)(sum + featN);
    __bf16* hbf   = xbf + featN;
    __bf16* aggbf = hbf + featN;

    const long long zero1 = (long long)N_NODES + featN;   // cnt + sum contiguous
    const long long scatW = (long long)E_EDGES * 16;
    const long long cvtW  = featN / 4;

    zero_kernel<<<(int)((zero1 + 255) / 256), 256, 0, stream>>>(cnt, zero1);
    degree_kernel<<<(E_EDGES + 255) / 256, 256, 0, stream>>>(dst, cnt, E_EDGES);
    convert_x_kernel<<<(int)((cvtW + 255) / 256), 256, 0, stream>>>(x, xbf, featN);
    scatter_bf16_kernel<<<(int)((scatW + 255) / 256), 256, 0, stream>>>(xbf, src, dst, sum, E_EDGES);
    convert_agg_kernel<<<(int)((cvtW + 255) / 256), 256, 0, stream>>>(sum, cnt, aggbf, featN);
    sage_layer1<<<BLOCKS_GEMM, 256, 0, stream>>>(aggbf, xbf, W1l, b1, W1r, hbf);
    zero_kernel<<<(int)((featN + 255) / 256), 256, 0, stream>>>(sum, featN);
    scatter_bf16_kernel<<<(int)((scatW + 255) / 256), 256, 0, stream>>>(hbf, src, dst, sum, E_EDGES);
    convert_agg_kernel<<<(int)((cvtW + 255) / 256), 256, 0, stream>>>(sum, cnt, aggbf, featN);
    sage_layer2<<<BLOCKS_GEMM, 256, 0, stream>>>(aggbf, hbf, W2l, b2, W2r, out);
}
